// VectorQuantizer_8186207666855
// MI455X (gfx1250) — matrix operand order repaired and checked by reading, not compiled, not run on hardware
//
#include <hip/hip_runtime.h>

#define N_EMB 1024
#define EDIM  256
#define HW    4096        // 64*64
#define NROWS 131072      // 32*4096
#define MROWS 32          // rows (z vectors) per block: 2 M-subtiles of 16

typedef __attribute__((ext_vector_type(16))) __bf16 v16bf;
typedef __attribute__((ext_vector_type(8)))  float  v8f;

union BfPack { uint4 q[2]; v16bf v; };

__device__ __forceinline__ unsigned short f2bf(float f) {
  union { float f; unsigned int u; } x; x.f = f;
  unsigned int r = x.u + 0x7FFFu + ((x.u >> 16) & 1u);   // round-to-nearest-even
  return (unsigned short)(r >> 16);
}

// ---------------- prep: emb fp32 -> bf16, e_sq = ||emb_row||^2 ----------------
__global__ __launch_bounds__(256) void vq_prep(const float* __restrict__ emb,
                                               unsigned short* __restrict__ emb_bf,
                                               float* __restrict__ e_sq) {
  int gwave = (blockIdx.x * 256 + threadIdx.x) >> 5;   // one wave per codebook row
  int lane  = threadIdx.x & 31;
  if (gwave >= N_EMB) return;
  const float* src = emb + (size_t)gwave * EDIM;
  float ss = 0.f;
#pragma unroll
  for (int i = 0; i < EDIM / 32; ++i) {
    int k = lane + i * 32;
    float v = src[k];
    ss += v * v;
    emb_bf[(size_t)gwave * EDIM + k] = f2bf(v);
  }
#pragma unroll
  for (int off = 16; off > 0; off >>= 1) ss += __shfl_down(ss, off, 32);
  if (lane == 0) e_sq[gwave] = ss;
}

// ---------------- main: WMMA GEMM (32xK x Kx1024) + argmin + gather ----------------
__global__ __launch_bounds__(256) void vq_main(const float* __restrict__ z,
                                               const float* __restrict__ emb,
                                               const unsigned short* __restrict__ emb_bf,
                                               const float* __restrict__ e_sq,
                                               float* __restrict__ zq,
                                               float* __restrict__ idx_out) {
  __shared__ __align__(16) unsigned short As[MROWS][264];  // bf16 z tile, padded rows
  __shared__ float esq_s[N_EMB];
  __shared__ float redv[8][MROWS][16];
  __shared__ int   redi[8][MROWS][16];
  __shared__ int   finalidx[MROWS];

  const int tid = threadIdx.x;
  const int m0  = blockIdx.x * MROWS;       // first flat row of this tile
  const int b   = m0 >> 12;                 // m0 / 4096 (MROWS divides HW -> same b)
  const int hw0 = m0 & (HW - 1);

  // ---- stage z tile as bf16 in A-friendly row-major LDS (32 rows x 256 K) ----
  {
    const int c = tid;                                         // channel 0..255
    const float* zsrc = z + ((size_t)b * EDIM + c) * HW + hw0; // 32 consecutive hw
#pragma unroll
    for (int r = 0; r < MROWS; ++r) As[r][c] = f2bf(zsrc[r]);
#pragma unroll
    for (int i = 0; i < N_EMB / 256; ++i) esq_s[tid + i * 256] = e_sq[tid + i * 256];
  }
  __syncthreads();

  const int wave  = tid >> 5;
  const int lane  = tid & 31;
  const int lrow  = lane & 15;   // A: M row / B,C: N column
  const int lhalf = lane >> 4;   // K half selector

  float bestv0[8], bestv1[8]; int besti0[8], besti1[8];
#pragma unroll
  for (int v = 0; v < 8; ++v) {
    bestv0[v] = 3.4e38f; besti0[v] = 0;
    bestv1[v] = 3.4e38f; besti1[v] = 0;
  }

  for (int nt = 0; nt < 8; ++nt) {
    const int col = wave * 128 + nt * 16 + lrow;              // codebook index
    v8f acc0 = {0.f, 0.f, 0.f, 0.f, 0.f, 0.f, 0.f, 0.f};
    v8f acc1 = {0.f, 0.f, 0.f, 0.f, 0.f, 0.f, 0.f, 0.f};
    const uint4* bptr = (const uint4*)(emb_bf + (size_t)col * EDIM + lhalf * 8);
#pragma unroll
    for (int ks = 0; ks < 8; ++ks) {                          // K = 256 in steps of 32
      BfPack a0, a1, bm;
      // B (32x16 bf16): lane = N, lanes 0-15 K=0-7/16-23, lanes 16-31 K=8-15/24-31
      bm.q[0] = bptr[ks * 4];
      bm.q[1] = bptr[ks * 4 + 2];
      // A (16x32 bf16): lanes 0-15 K=0-7/16-23, lanes 16-31 K=8-15/24-31
      a0.q[0] = *(const uint4*)&As[lrow][ks * 32 + lhalf * 8];
      a0.q[1] = *(const uint4*)&As[lrow][ks * 32 + lhalf * 8 + 16];
      a1.q[0] = *(const uint4*)&As[lrow + 16][ks * 32 + lhalf * 8];
      a1.q[1] = *(const uint4*)&As[lrow + 16][ks * 32 + lhalf * 8 + 16];
      // one B feeds two M-subtiles: halves L2 traffic per WMMA
      acc0 = __builtin_amdgcn_wmma_f32_16x16x32_bf16(false, a0.v, false, bm.v,
                                                     (short)0, acc0, false, false);
      acc1 = __builtin_amdgcn_wmma_f32_16x16x32_bf16(false, a1.v, false, bm.v,
                                                     (short)0, acc1, false, false);
    }
    const float es = esq_s[col];
#pragma unroll
    for (int v = 0; v < 8; ++v) {                 // score = ||e||^2 - 2*dot
      float s0 = es - 2.0f * acc0[v];
      if (s0 < bestv0[v]) { bestv0[v] = s0; besti0[v] = col; }
      float s1 = es - 2.0f * acc1[v];
      if (s1 < bestv1[v]) { bestv1[v] = s1; besti1[v] = col; }
    }
  }

  // C layout: VGPR v -> row v + 8*lhalf, lane&15 -> column slot
#pragma unroll
  for (int v = 0; v < 8; ++v) {
    redv[wave][v + lhalf * 8][lrow]      = bestv0[v];
    redi[wave][v + lhalf * 8][lrow]      = besti0[v];
    redv[wave][16 + v + lhalf * 8][lrow] = bestv1[v];
    redi[wave][16 + v + lhalf * 8][lrow] = besti1[v];
  }
  __syncthreads();

  if (tid < MROWS) {                              // final argmin over 8 waves x 16 slots
    float bv = 3.4e38f; int bi = 0;
    for (int w = 0; w < 8; ++w)
#pragma unroll
      for (int c2 = 0; c2 < 16; ++c2) {
        float v = redv[w][tid][c2];
        if (v < bv) { bv = v; bi = redi[w][tid][c2]; }
      }
    finalidx[tid] = bi;
    idx_out[m0 + tid] = (float)bi;
  }
  __syncthreads();

  // gather z_q = emb[idx] back into [b, c, h, w]; thread = channel, 32 hw each
  {
    const int c = tid;
    const size_t obase = ((size_t)b * EDIM + c) * HW + hw0;
#pragma unroll
    for (int r = 0; r < MROWS; ++r)
      zq[obase + r] = emb[(size_t)finalidx[r] * EDIM + c];
  }
}

extern "C" void kernel_launch(void* const* d_in, const int* in_sizes, int n_in,
                              void* d_out, int out_size, void* d_ws, size_t ws_size,
                              hipStream_t stream) {
  (void)in_sizes; (void)n_in; (void)out_size; (void)ws_size;
  const float* z   = (const float*)d_in[0];
  const float* emb = (const float*)d_in[1];

  float*          e_sq   = (float*)d_ws;                              // 4 KB
  unsigned short* emb_bf = (unsigned short*)((char*)d_ws + 4096);     // 512 KB

  float* zq      = (float*)d_out;
  float* idx_out = zq + (size_t)32 * EDIM * HW;                       // after 33,554,432 floats

  vq_prep<<<N_EMB / 8, 256, 0, stream>>>(emb, emb_bf, e_sq);
  vq_main<<<NROWS / MROWS, 256, 0, stream>>>(z, emb, emb_bf, e_sq, zq, idx_out);
}
